// Encoder1_79087527789130
// MI455X (gfx1250) — matrix operand order repaired and checked by reading, not compiled, not run on hardware
//
#include <hip/hip_runtime.h>
#include <math.h>

typedef __attribute__((ext_vector_type(16))) _Float16 v16h;
typedef __attribute__((ext_vector_type(8)))  _Float16 v8h;
typedef __attribute__((ext_vector_type(8)))  float    v8f;

#define EPSB 1e-5f

__device__ __forceinline__ v8f wmma_f16(v16h a, v16h b, v8f c) {
  return __builtin_amdgcn_wmma_f32_16x16x32_f16(false, a, false, b, (short)0, c, false, false);
}

// A-matrix 16x32 f16 fragment from row-major LDS (ISA 7.12.2), two b128 loads:
// lane<16:  halves 0-7 = K 0..7,  halves 8-15 = K 16..23 (row M = lane&15)
// lane>=16: halves 0-7 = K 8..15, halves 8-15 = K 24..31
__device__ __forceinline__ v16h a_frag_lds(const _Float16* act, int ld, int m0, int k0, int lane) {
  const _Float16* row = act + (m0 + (lane & 15)) * ld + k0 + 8 * (lane >> 4);
  v8h lo = *(const v8h*)row;
  v8h hh = *(const v8h*)(row + 16);
  return __builtin_shufflevector(lo, hh, 0, 1, 2, 3, 4, 5, 6, 7, 8, 9, 10, 11, 12, 13, 14, 15);
}

// B-matrix fragment from pre-swizzled f16 weights: one aligned 32B load.
// layout: dst[((tile*32 + lane)*16 + h] = W[kt*32 + 8*(lane>>4) + (h&7) + 16*(h>>3)][nt*16 + (lane&15)]
__device__ __forceinline__ v16h b_frag_swz(const _Float16* w, int tile, int lane) {
  return *(const v16h*)(w + (tile * 32 + lane) * 16);
}

__device__ __forceinline__ float sigmoidf_(float x) {
  return __builtin_amdgcn_rcpf(1.0f + __expf(-x));
}

// ---- weight pre-swizzle into B-fragment layout (f32 -> f16), one-time ----
__global__ __launch_bounds__(256) void swz_kernel(const float* __restrict__ W, int K, int N,
                                                  _Float16* __restrict__ dst) {
  int idx = blockIdx.x * 256 + threadIdx.x;
  if (idx >= K * N) return;
  int h = idx & 15;
  int l = (idx >> 4) & 31;
  int tile = idx >> 9;
  int ntc = N >> 4;
  int kt = tile / ntc, nt = tile - kt * ntc;
  int krow = kt * 32 + 8 * (l >> 4) + (h & 7) + 16 * (h >> 3);
  int ncol = nt * 16 + (l & 15);
  dst[idx] = (_Float16)W[krow * N + ncol];
}

__global__ __launch_bounds__(256) void swz_sum_kernel(const float* __restrict__ Wa,
                                                      const float* __restrict__ Wb, int K, int N,
                                                      _Float16* __restrict__ dst) {
  int idx = blockIdx.x * 256 + threadIdx.x;
  if (idx >= K * N) return;
  int h = idx & 15;
  int l = (idx >> 4) & 31;
  int tile = idx >> 9;
  int ntc = N >> 4;
  int kt = tile / ntc, nt = tile - kt * ntc;
  int e = (kt * 32 + 8 * (l >> 4) + (h & 7) + 16 * (h >> 3)) * N + nt * 16 + (l & 15);
  dst[idx] = (_Float16)(Wa[e] + Wb[e]);
}

// GEMM layer: act_out = relu(BN(act_in @ W + b)), 128 rows, LDS f16 in/out.
__device__ __forceinline__ void mlp_gemm_layer(const _Float16* actin, int K, _Float16* actout,
                                               int N, const _Float16* Wswz, const float* cS,
                                               const float* cB, int wave, int lane) {
  int ntc = N >> 4;
  int KT = K >> 5;
  int hi = lane >> 4, nn = lane & 15;
  for (int t = wave; t < 8 * ntc; t += 8) {
    int m0 = (t / ntc) << 4;
    int nt = t % ntc;
    v8f acc = {};
    for (int k = 0; k < KT; ++k)
      acc = wmma_f16(a_frag_lds(actin, K, m0, k * 32, lane), b_frag_swz(Wswz, k * ntc + nt, lane),
                     acc);
    int n = (nt << 4) + nn;
    float s = cS[n], o = cB[n];
#pragma unroll
    for (int r = 0; r < 8; ++r) {
      float v = acc[r] * s + o;
      actout[(m0 + r + 8 * hi) * N + n] = (_Float16)fmaxf(v, 0.0f);
    }
  }
}

__device__ __forceinline__ void chan_params(int tid, int N, const float* g, const float* bias,
                                            const float* beta, float* cS, float* cB) {
  if (tid < N) {
    float s = g[tid] * rsqrtf(1.0f + EPSB);
    cS[tid] = s;
    cB[tid] = bias[tid] * s + beta[tid];
  }
}

// ---------------- NNConv 1: node_feat(2048,7) -> x1(2048,64) ----------------
__global__ __launch_bounds__(256) void nnconv_mlp64_kernel(
    const float* __restrict__ edge_feat, const float* __restrict__ node_feat,
    const int* __restrict__ src, const float* W0, const _Float16* __restrict__ W1s,
    const _Float16* __restrict__ W2s, const _Float16* __restrict__ W3s,
    const _Float16* __restrict__ W4s, const float* b0, const float* b1, const float* b2,
    const float* b3, const float* b4, const float* g0, const float* g1, const float* g2,
    const float* g3, const float* be0, const float* be1, const float* be2, const float* be3,
    const float* __restrict__ bias1, float* __restrict__ x1) {
  __shared__ __align__(16) char smem[65536 + 32768];  // actA(64K) | actB(32K)
  __shared__ float cS[256], cB[256], ef[384];
  _Float16* actA = (_Float16*)smem;
  _Float16* actB = (_Float16*)(smem + 65536);
  float* hsrc7 = (float*)smem;             // overlay after L3: 128*7*4
  float* b4l   = (float*)(smem + 4096);    // 448*4

  int tid = threadIdx.x, lane = tid & 31, wave = tid >> 5;
  int e0 = blockIdx.x * 128;

  for (int i = tid; i < 384; i += 256) ef[i] = edge_feat[e0 * 3 + i];
  chan_params(tid, 256, g0, b0, be0, cS, cB);
  __syncthreads();
  for (int idx = tid; idx < 128 * 256; idx += 256) {
    int row = idx >> 8, n = idx & 255;
    float v = ef[row * 3] * W0[n] + ef[row * 3 + 1] * W0[256 + n] + ef[row * 3 + 2] * W0[512 + n];
    actA[idx] = (_Float16)fmaxf(v * cS[n] + cB[n], 0.0f);
  }
  __syncthreads();
  chan_params(tid, 128, g1, b1, be1, cS, cB);
  __syncthreads();
  mlp_gemm_layer(actA, 256, actB, 128, W1s, cS, cB, wave, lane);
  __syncthreads();
  chan_params(tid, 64, g2, b2, be2, cS, cB);
  __syncthreads();
  mlp_gemm_layer(actB, 128, actA, 64, W2s, cS, cB, wave, lane);
  __syncthreads();
  chan_params(tid, 32, g3, b3, be3, cS, cB);
  __syncthreads();
  mlp_gemm_layer(actA, 64, actB, 32, W3s, cS, cB, wave, lane);  // Z -> actB
  __syncthreads();
  // actA dead: gather node feats + stage b4
  for (int idx = tid; idx < 128 * 7; idx += 256) {
    int row = idx / 7, i = idx - row * 7;
    hsrc7[idx] = node_feat[src[e0 + row] * 7 + i];
  }
  for (int idx = tid; idx < 448; idx += 256) b4l[idx] = b4[idx];
  __syncthreads();
  // fused: msg[m][o] = sum_i h[m][i] * sigmoid((Z@W4)[m][i*64+o] + b4[i*64+o])
  v16h za = a_frag_lds(actB, 32, wave * 16, 0, lane);
  int hi = lane >> 4, nn = lane & 15;
  v8f acc[4] = {};
#pragma unroll 4
  for (int cb = 0; cb < 28; ++cb) {
    int i = cb >> 2, og = cb & 3;
    v8f z = {};
    v8f d = wmma_f16(za, b_frag_swz(W4s, cb, lane), z);
    float bb = b4l[cb * 16 + nn];
    float coef[8];
#pragma unroll
    for (int r = 0; r < 8; ++r) coef[r] = hsrc7[(wave * 16 + r + 8 * hi) * 7 + i];
#pragma unroll
    for (int r = 0; r < 8; ++r) acc[og][r] += coef[r] * sigmoidf_(d[r] + bb);
  }
  // 8 contiguous edges per node -> mean + bias
  int node = blockIdx.x * 16 + wave * 2 + hi;
#pragma unroll
  for (int og = 0; og < 4; ++og) {
    float s = 0.0f;
#pragma unroll
    for (int r = 0; r < 8; ++r) s += acc[og][r];
    int o = og * 16 + nn;
    x1[node * 64 + o] = s * 0.125f + bias1[o];
  }
}

// ---------------- NNConv 2: x1(2048,64) -> x2(2048,128) ----------------
__global__ __launch_bounds__(256) void nnconv_mlp128_kernel(
    const float* __restrict__ edge_feat, const float* __restrict__ x1in,
    const int* __restrict__ src, const float* W0, const _Float16* __restrict__ W1s,
    const _Float16* __restrict__ W2s, const _Float16* __restrict__ W3s,
    const _Float16* __restrict__ W4s, const float* b0, const float* b1, const float* b2,
    const float* b3, const float* b4, const float* g0, const float* g1, const float* g2,
    const float* g3, const float* be0, const float* be1, const float* be2, const float* be3,
    const float* __restrict__ bias2, float* __restrict__ x2) {
  __shared__ __align__(16) char smem[65536 + 32768];
  __shared__ float cS[256], cB[256], ef[384];
  _Float16* actA = (_Float16*)smem;
  _Float16* actB = (_Float16*)(smem + 65536);
  float* hsrc = (float*)smem;              // overlay after L3: 128*64*4 = 32768
  float* b4l  = (float*)(smem + 32768);    // 8192*4 = 32768

  int tid = threadIdx.x, lane = tid & 31, wave = tid >> 5;
  int e0 = blockIdx.x * 128;

  for (int i = tid; i < 384; i += 256) ef[i] = edge_feat[e0 * 3 + i];
  chan_params(tid, 256, g0, b0, be0, cS, cB);
  __syncthreads();
  for (int idx = tid; idx < 128 * 256; idx += 256) {
    int row = idx >> 8, n = idx & 255;
    float v = ef[row * 3] * W0[n] + ef[row * 3 + 1] * W0[256 + n] + ef[row * 3 + 2] * W0[512 + n];
    actA[idx] = (_Float16)fmaxf(v * cS[n] + cB[n], 0.0f);
  }
  __syncthreads();
  chan_params(tid, 128, g1, b1, be1, cS, cB);
  __syncthreads();
  mlp_gemm_layer(actA, 256, actB, 128, W1s, cS, cB, wave, lane);
  __syncthreads();
  chan_params(tid, 64, g2, b2, be2, cS, cB);
  __syncthreads();
  mlp_gemm_layer(actB, 128, actA, 64, W2s, cS, cB, wave, lane);
  __syncthreads();
  chan_params(tid, 32, g3, b3, be3, cS, cB);
  __syncthreads();
  mlp_gemm_layer(actA, 64, actB, 32, W3s, cS, cB, wave, lane);  // Z -> actB
  __syncthreads();
  for (int idx = tid; idx < 128 * 64; idx += 256) {
    int row = idx >> 6;
    hsrc[idx] = x1in[src[e0 + row] * 64 + (idx & 63)];
  }
  for (int idx = tid; idx < 8192; idx += 256) b4l[idx] = b4[idx];
  __syncthreads();
  v16h za = a_frag_lds(actB, 32, wave * 16, 0, lane);
  int hi = lane >> 4, nn = lane & 15;
  v8f acc[8] = {};
  // chunk = one i-slice of the (64,128) per-edge weight; tiles stream from L2.
  for (int chunk = 0; chunk < 64; ++chunk) {
    float coef[8];
#pragma unroll
    for (int r = 0; r < 8; ++r) coef[r] = hsrc[(wave * 16 + r + 8 * hi) * 64 + chunk];
#pragma unroll
    for (int cb = 0; cb < 8; ++cb) {
      int t = chunk * 8 + cb;
      v8f z = {};
      v8f d = wmma_f16(za, b_frag_swz(W4s, t, lane), z);
      float bb = b4l[t * 16 + nn];
#pragma unroll
      for (int r = 0; r < 8; ++r) acc[cb][r] += coef[r] * sigmoidf_(d[r] + bb);
    }
  }
  int node = blockIdx.x * 16 + wave * 2 + hi;
#pragma unroll
  for (int cb = 0; cb < 8; ++cb) {
    float s = 0.0f;
#pragma unroll
    for (int r = 0; r < 8; ++r) s += acc[cb][r];
    int o = cb * 16 + nn;
    x2[node * 128 + o] = s * 0.125f + bias2[o];
  }
}

// ---------------- EdgeConv: P = X@(Wt+Wp)+bt+bp, Q = X@Wt ----------------
template <int IN, int OUT>
__global__ __launch_bounds__(256) void edgeconv_gemm_kernel(
    const float* __restrict__ X, const _Float16* __restrict__ Wq, const _Float16* __restrict__ Wps,
    const float* __restrict__ bt, const float* __restrict__ bp, float* __restrict__ P,
    float* __restrict__ Q) {
  __shared__ _Float16 xh[128 * IN];
  int tid = threadIdx.x, lane = tid & 31, wave = tid >> 5;
  int r0 = blockIdx.x * 128;
  for (int idx = tid; idx < 128 * IN; idx += 256) xh[idx] = (_Float16)X[r0 * IN + idx];
  __syncthreads();
  const int NT = OUT / 16, KT = IN / 32;
  int hi = lane >> 4, nn = lane & 15;
  for (int t = wave; t < 8 * NT * 2; t += 8) {
    int which = t & 1, tt = t >> 1;
    int m0 = (tt / NT) * 16, nt = tt % NT;
    const _Float16* W = which ? Wps : Wq;
    v8f acc = {};
    for (int k = 0; k < KT; ++k)
      acc = wmma_f16(a_frag_lds(xh, IN, m0, k * 32, lane), b_frag_swz(W, k * NT + nt, lane), acc);
    int n = nt * 16 + nn;
    if (which) {
      float bias = bt[n] + bp[n];
#pragma unroll
      for (int r = 0; r < 8; ++r) P[(r0 + m0 + r + 8 * hi) * OUT + n] = acc[r] + bias;
    } else {
#pragma unroll
      for (int r = 0; r < 8; ++r) Q[(r0 + m0 + r + 8 * hi) * OUT + n] = acc[r];
    }
  }
}

// y[v][o] = max_{j<8} BN(P[v][o] - Q[src[8v+j]][o])
template <int OUT>
__global__ __launch_bounds__(256) void edgeconv_reduce_kernel(
    const float* __restrict__ P, const float* __restrict__ Q, const int* __restrict__ src,
    const float* __restrict__ g, const float* __restrict__ beta, float* __restrict__ Y) {
  int idx = blockIdx.x * 256 + threadIdx.x;
  if (idx >= 2048 * OUT) return;
  int v = idx / OUT, o = idx - v * OUT;
  float s = g[o] * rsqrtf(1.0f + EPSB);
  float be = beta[o];
  float p = P[idx];
  float m = -3.4e38f;
#pragma unroll
  for (int j = 0; j < 8; ++j) {
    int sv = src[v * 8 + j];
    m = fmaxf(m, (p - Q[sv * OUT + o]) * s + be);
  }
  Y[idx] = m;
}

static inline void swz(const float* W, int K, int N, _Float16* dst, hipStream_t s) {
  swz_kernel<<<(K * N + 255) / 256, 256, 0, s>>>(W, K, N, dst);
}
static inline void swzsum(const float* Wa, const float* Wb, int K, int N, _Float16* dst,
                          hipStream_t s) {
  swz_sum_kernel<<<(K * N + 255) / 256, 256, 0, s>>>(Wa, Wb, K, N, dst);
}

extern "C" void kernel_launch(void* const* d_in, const int* in_sizes, int n_in, void* d_out,
                              int out_size, void* d_ws, size_t ws_size, hipStream_t stream) {
  (void)in_sizes; (void)n_in; (void)out_size; (void)ws_size;
  const float* node_feat = (const float*)d_in[0];
  const float* edge_feat = (const float*)d_in[1];
  const float* A[18]; for (int i = 0; i < 18; ++i) A[i] = (const float*)d_in[2 + i];
  const float* B[18]; for (int i = 0; i < 18; ++i) B[i] = (const float*)d_in[20 + i];
  const float* bias1 = (const float*)d_in[38];
  const float* bias2 = (const float*)d_in[39];
  const float* E1[6]; for (int i = 0; i < 6; ++i) E1[i] = (const float*)d_in[40 + i];
  const float* E2[6]; for (int i = 0; i < 6; ++i) E2[i] = (const float*)d_in[46 + i];
  const float* E3[6]; for (int i = 0; i < 6; ++i) E3[i] = (const float*)d_in[52 + i];
  const int* src = (const int*)d_in[58];

  char* ws = (char*)d_ws;
  size_t cur = 0;
  auto allocF = [&](size_t n) { float* p = (float*)(ws + cur); cur += n * 4; return p; };
  auto allocH = [&](size_t n) {
    cur = (cur + 31) & ~(size_t)31;
    _Float16* p = (_Float16*)(ws + cur);
    cur += n * 2;
    return p;
  };
  float* x1 = allocF(2048 * 64);
  float* x2 = allocF(2048 * 128);
  float* P  = allocF(2048 * 64);
  float* Q  = allocF(2048 * 64);
  float* h1 = allocF(2048 * 64);
  float* h2 = allocF(2048 * 32);
  _Float16* w1a = allocH(256 * 128);
  _Float16* w2a = allocH(128 * 64);
  _Float16* w3a = allocH(64 * 32);
  _Float16* w4a = allocH(32 * 448);
  _Float16* w1b = allocH(256 * 128);
  _Float16* w2b = allocH(128 * 64);
  _Float16* w3b = allocH(64 * 32);
  _Float16* w4b = allocH(32 * 8192);
  _Float16* e1q = allocH(128 * 64);
  _Float16* e1p = allocH(128 * 64);
  _Float16* e2q = allocH(64 * 32);
  _Float16* e2p = allocH(64 * 32);
  _Float16* e3q = allocH(32 * 16);
  _Float16* e3p = allocH(32 * 16);
  float* out = (float*)d_out;

  // one-time (per launch) weight swizzles, all independent
  swz(A[1], 256, 128, w1a, stream);
  swz(A[2], 128, 64, w2a, stream);
  swz(A[3], 64, 32, w3a, stream);
  swz(A[4], 32, 448, w4a, stream);
  swz(B[1], 256, 128, w1b, stream);
  swz(B[2], 128, 64, w2b, stream);
  swz(B[3], 64, 32, w3b, stream);
  swz(B[4], 32, 8192, w4b, stream);
  swz(E1[0], 128, 64, e1q, stream);
  swzsum(E1[0], E1[2], 128, 64, e1p, stream);
  swz(E2[0], 64, 32, e2q, stream);
  swzsum(E2[0], E2[2], 64, 32, e2p, stream);
  swz(E3[0], 32, 16, e3q, stream);
  swzsum(E3[0], E3[2], 32, 16, e3p, stream);

  nnconv_mlp64_kernel<<<128, 256, 0, stream>>>(
      edge_feat, node_feat, src, A[0], w1a, w2a, w3a, w4a, A[5], A[6], A[7], A[8], A[9], A[10],
      A[11], A[12], A[13], A[14], A[15], A[16], A[17], bias1, x1);
  nnconv_mlp128_kernel<<<128, 256, 0, stream>>>(
      edge_feat, x1, src, B[0], w1b, w2b, w3b, w4b, B[5], B[6], B[7], B[8], B[9], B[10], B[11],
      B[12], B[13], B[14], B[15], B[16], B[17], bias2, x2);

  edgeconv_gemm_kernel<128, 64><<<16, 256, 0, stream>>>(x2, e1q, e1p, E1[1], E1[3], P, Q);
  edgeconv_reduce_kernel<64><<<(2048 * 64) / 256, 256, 0, stream>>>(P, Q, src, E1[4], E1[5], h1);
  edgeconv_gemm_kernel<64, 32><<<16, 256, 0, stream>>>(h1, e2q, e2p, E2[1], E2[3], P, Q);
  edgeconv_reduce_kernel<32><<<(2048 * 32) / 256, 256, 0, stream>>>(P, Q, src, E2[4], E2[5], h2);
  edgeconv_gemm_kernel<32, 16><<<16, 256, 0, stream>>>(h2, e3q, e3p, E3[1], E3[3], P, Q);
  edgeconv_reduce_kernel<16><<<(2048 * 16) / 256, 256, 0, stream>>>(P, Q, src, E3[4], E3[5], out);
}
